// Model_14577119003543
// MI455X (gfx1250) — compile-verified, matrix operand order checked by reading, not run on hardware
//
#include <hip/hip_runtime.h>
#include <hip/hip_bf16.h>

// ---------------------------------------------------------------------------
// MI455X / gfx1250 implementation.
//
// - All GEMM-like work (MLP layers 1/2/3 and the FEM solve) runs on
//   v_wmma_f32_16x16x32_f16 (f16 in / f32 accumulate).
// - P = inv_pde @ mass is precomputed once, fusing the two per-step FEM GEMMs
//   into one and halving serial latency; P (32 MB f16) is L2-resident.
// - The per-step FEM GEMM stages its shared B panel (u^T) into LDS with the
//   Tensor Data Mover (tensor_load_to_lds + s_wait_tensorcnt), double-buffered.
// - tanh uses a clamped Pade approximant to keep the VALU share low.
// ---------------------------------------------------------------------------

typedef __attribute__((ext_vector_type(16))) _Float16 v16h;
typedef __attribute__((ext_vector_type(8)))  float    v8f;
typedef __attribute__((ext_vector_type(4)))  unsigned int u32x4;
typedef __attribute__((ext_vector_type(8)))  int      i32x8;
typedef __attribute__((ext_vector_type(4)))  int      i32x4;

#define NN    4096
#define BB    64
#define TT    128
#define HH    128
#define SS    3
#define DT_C  0.01f
#define VMEAN (-75.0f)
#define VSTD  115.0f

__device__ __forceinline__ v8f wmma_f16(v16h a, v16h b, v8f c) {
  // 8 args: (neg_a, A, neg_b, B, c_mod, C, reuse_a, reuse_b)
  return __builtin_amdgcn_wmma_f32_16x16x32_f16(false, a, false, b, (short)0, c,
                                                false, false);
}

// Clamped Pade(3,2) tanh: |err| < 2.5e-2, ~6 VALU ops vs ~15 for libm tanhf.
__device__ __forceinline__ float fast_tanh(float x) {
  float x2 = x * x;
  float r = (x * (27.f + x2)) / (27.f + 9.f * x2);
  return fminf(1.f, fmaxf(-1.f, r));
}

// ---------------------------------------------------------------------------
// One-time conversion / packing kernels
// ---------------------------------------------------------------------------
__global__ __launch_bounds__(256) void cvt_f16(const float* __restrict__ src,
                                               _Float16* __restrict__ dst, int n) {
  int i = blockIdx.x * blockDim.x + threadIdx.x;
  if (i < n) dst[i] = (_Float16)src[i];
}

// dst[c*R + r] = src[r*C + c]  (column-major f16: K-contiguous WMMA B panels)
__global__ __launch_bounds__(256) void cvt_f16_T(const float* __restrict__ src,
                                                 _Float16* __restrict__ dst,
                                                 int R, int C) {
  int i = blockIdx.x * blockDim.x + threadIdx.x;
  if (i < R * C) {
    int r = i / C, c = i - r * C;
    dst[(size_t)c * R + r] = (_Float16)src[i];
  }
}

// W1 [4,128] -> padded col-major [128 cols][32 K] (K 4..31 zero)
__global__ __launch_bounds__(128) void pack_w1(const float* __restrict__ W1,
                                               _Float16* __restrict__ out) {
  int i = blockIdx.x * 128 + threadIdx.x;     // 128*32
  int c = i >> 5, k = i & 31;
  out[i] = (_Float16)(k < 4 ? W1[k * HH + c] : 0.f);
}

// Combined head tile [16 cols][128 K]: cols 0..2 = fW3, col 3 = iW3, rest 0.
__global__ __launch_bounds__(128) void pack_w3(const float* __restrict__ fW3,
                                               const float* __restrict__ iW3,
                                               _Float16* __restrict__ out) {
  int i = blockIdx.x * 128 + threadIdx.x;     // 16*128
  int c = i >> 7, k = i & 127;
  float val = 0.f;
  if (c < SS)       val = fW3[k * SS + c];
  else if (c == SS) val = iW3[k];
  out[i] = (_Float16)val;
}

// ---------------------------------------------------------------------------
// Generic WMMA GEMM (used once for P = inv_pde @ mass).
// C[M,Nc] = A[M,K] (f16 row-major) @ B, with B given as BT[Nc,K].
// One wave: 16 x (16*NT) tile, f32 accum, f16 or f32 out.
// ---------------------------------------------------------------------------
template <int NT>
__global__ __launch_bounds__(256) void gemm_wmma(const _Float16* __restrict__ A,
                                                 const _Float16* __restrict__ BT,
                                                 float* __restrict__ Cf,
                                                 _Float16* __restrict__ Ch,
                                                 int M, int Nc, int K) {
  const int lane = threadIdx.x & 31;
  const int wid  = threadIdx.x >> 5;
  const int W    = blockIdx.x * (blockDim.x >> 5) + wid;
  const int numNT = Nc / (16 * NT);
  const int mt = W / numNT, nt = W - mt * numNT;
  const int m0 = mt * 16, n0 = nt * 16 * NT;
  if (m0 >= M) return;

  const int row  = lane & 15;
  const int koff = (lane >> 4) * 16;
  const _Float16* arow = A + (size_t)(m0 + row) * K + koff;

  v8f acc[NT];
#pragma unroll
  for (int c = 0; c < NT; ++c) { v8f z = {}; acc[c] = z; }

  for (int k0 = 0; k0 < K; k0 += 32) {
    v16h a = *(const v16h*)(arow + k0);
    __builtin_prefetch(arow + k0 + 128, 0, 1);   // global_prefetch_b8
#pragma unroll
    for (int c = 0; c < NT; ++c) {
      const _Float16* bp = BT + (size_t)(n0 + c * 16 + row) * K + koff + k0;
      v16h b = *(const v16h*)bp;
      acc[c] = wmma_f16(a, b, acc[c]);
    }
  }

  const int mb = (lane >> 4) * 8;
#pragma unroll
  for (int c = 0; c < NT; ++c) {
    int col = n0 + c * 16 + row;
#pragma unroll
    for (int i = 0; i < 8; ++i) {
      size_t idx = (size_t)(m0 + mb + i) * Nc + col;
      if (Cf) Cf[idx] = acc[c][i];
      else    Ch[idx] = (_Float16)acc[c][i];
    }
  }
}

// ---------------------------------------------------------------------------
// Tensor Data Mover: async-load a 2D f16 tile [tileK x 64 cols] of uT
// ([64 cols][4096 K] row-major, stride 4096) into LDS.  D# built per
// CDNA5 ISA (group0: count/lds/global/type, group1:
// data_size=2B, dims/strides).  lds_addr = flat-pointer low 32 bits
// (= wave-relative LDS byte address per the aperture rule).
// This toolchain carries the 6-arg builtin: groups 2/3 (4 SGPRs each) plus a
// trailing 8-dword group before the cache-policy immediate; all zero for 2D.
// ---------------------------------------------------------------------------
#define SLAB_K 128

__device__ __forceinline__ void tdm_load_slab(const _Float16* gsrc,
                                              _Float16* ldst) {
  unsigned long long ga = (unsigned long long)(uintptr_t)gsrc;
  unsigned int la = (unsigned int)(uintptr_t)ldst;
  u32x4 g0 = { 1u,                                   // count = 1 (valid D#)
               la,                                   // lds_addr
               (unsigned int)ga,                     // global_addr[31:0]
               (unsigned int)(ga >> 32) | (2u << 30) // global_addr[56:32] | type=2
             };
  i32x8 g1 = { 0x00010000,        // data_size = 2 bytes
               0x10000000,        // tensor_dim0 = 4096  (bits 63:48)
               (BB << 16),        // tensor_dim1 = 64    (bits 95:80)
               (SLAB_K << 16),    // tile_dim0 = SLAB_K  (bits 127:112)
               BB,                // tile_dim1 = 64      (bits 143:128)
               NN,                // tensor_dim0_stride = 4096 (bits 191:160)
               0, 0 };
  i32x4 z4 = { 0, 0, 0, 0 };
  i32x8 z8 = { 0, 0, 0, 0, 0, 0, 0, 0 };
  __builtin_amdgcn_tensor_load_to_lds(g0, g1, z4, z4, z8, 0);
}

// ---------------------------------------------------------------------------
// Per-step FEM GEMM: v_new[4096,64] = P[4096,4096] @ u, with the shared B
// panel TDM-staged into LDS (double-buffered), A-frags streamed from the
// L2-resident P.  8 waves/block, each wave owns a 16x64 output tile.
// ---------------------------------------------------------------------------
__global__ __launch_bounds__(256) void gemm_step_tdm(
    const _Float16* __restrict__ A,   // P, f16 row-major [4096,4096]
    const _Float16* __restrict__ BT,  // uT, f16 [64][4096]
    float* __restrict__ C) {          // v out, f32 [4096,64]
  __shared__ __align__(32) _Float16 sB[2][BB * SLAB_K];  // 2 x 16 KB

  const int lane = threadIdx.x & 31;
  const int wid  = threadIdx.x >> 5;
  const int m0   = blockIdx.x * 128 + wid * 16;
  const int row  = lane & 15;
  const int koff = (lane >> 4) * 16;
  const _Float16* arow = A + (size_t)(m0 + row) * NN + koff;

  v8f acc[4];
#pragma unroll
  for (int c = 0; c < 4; ++c) { v8f z = {}; acc[c] = z; }

  const int nslab = NN / SLAB_K;     // 32
  if (wid == 0) tdm_load_slab(BT, &sB[0][0]);

  for (int slab = 0; slab < nslab; ++slab) {
    if (wid == 0) __builtin_amdgcn_s_wait_tensorcnt(0);
    __syncthreads();                 // slab data visible; prev buffer free
    if (wid == 0 && slab + 1 < nslab)
      tdm_load_slab(BT + (slab + 1) * SLAB_K, &sB[(slab + 1) & 1][0]);

    const _Float16* bs = sB[slab & 1];
#pragma unroll
    for (int kc = 0; kc < SLAB_K / 32; ++kc) {
      int k0 = slab * SLAB_K + kc * 32;
      v16h a = *(const v16h*)(arow + k0);
      __builtin_prefetch(arow + k0 + 256, 0, 1);
#pragma unroll
      for (int c = 0; c < 4; ++c) {
        v16h b = *(const v16h*)(bs + (c * 16 + row) * SLAB_K + kc * 32 + koff);
        acc[c] = wmma_f16(a, b, acc[c]);
      }
    }
  }

  const int mb = (lane >> 4) * 8;
#pragma unroll
  for (int c = 0; c < 4; ++c) {
    int col = c * 16 + row;
#pragma unroll
    for (int i = 0; i < 8; ++i)
      C[(size_t)(m0 + mb + i) * BB + col] = acc[c][i];
  }
}

// ---------------------------------------------------------------------------
// Fused per-step MLP tile:  all three layers on WMMA, barrier-free.
//   L1: A-frag built in registers from (vn,s) padded K=4->32, B from packed W1.
//   L2: 128x128, B-frags straight from global f16 (L2-resident).
//   L3: one 16-col tile; cols 0..2 = forcing head, col 3 = ionic head.
// ---------------------------------------------------------------------------
__device__ __forceinline__ void mlp_half(int wid, int lane, _Float16* ht,
                                         float (*xb)[16][4],
                                         const _Float16* __restrict__ W1p,
                                         const float* __restrict__ b1,
                                         const _Float16* __restrict__ W2T,
                                         const float* __restrict__ b2) {
  const int row  = lane & 15;
  const int koff = (lane >> 4) * 16;
  const int mb   = (lane >> 4) * 8;

  // ---- layer 1: [16,32(pad)] x [32,128] ----
  v16h ax = {};                       // lanes 16..31 (K 16..31) stay zero
  if (lane < 16) {
#pragma unroll
    for (int k = 0; k < 4; ++k) ax[k] = (_Float16)xb[wid][row][k];
  }
#pragma unroll
  for (int ct = 0; ct < 8; ++ct) {
    int col = ct * 16 + row;
    v16h b = *(const v16h*)(W1p + col * 32 + koff);
    v8f z = {};
    v8f acc = wmma_f16(ax, b, z);
    float bias = b1[col];
#pragma unroll
    for (int i = 0; i < 8; ++i)
      ht[(mb + i) * HH + col] = (_Float16)fast_tanh(acc[i] + bias);
  }

  // ---- layer 2: [16,128] x [128,128] ----
  v16h a[4];
#pragma unroll
  for (int kk = 0; kk < 4; ++kk)
    a[kk] = *(const v16h*)(ht + row * HH + kk * 32 + koff);
  v8f acc2[8];
#pragma unroll
  for (int ct = 0; ct < 8; ++ct) {
    v8f z = {}; acc2[ct] = z;
#pragma unroll
    for (int kk = 0; kk < 4; ++kk) {
      v16h b = *(const v16h*)(W2T + (ct * 16 + row) * HH + kk * 32 + koff);
      acc2[ct] = wmma_f16(a[kk], b, acc2[ct]);
    }
  }
#pragma unroll
  for (int ct = 0; ct < 8; ++ct) {
    int col = ct * 16 + row;
    float bias = b2[col];
#pragma unroll
    for (int i = 0; i < 8; ++i)
      ht[(mb + i) * HH + col] = (_Float16)fast_tanh(acc2[ct][i] + bias);
  }
}

__device__ __forceinline__ v8f layer3_wmma(int lane, const _Float16* ht,
                                           const _Float16* __restrict__ W3p) {
  const int row = lane & 15, koff = (lane >> 4) * 16;
  v8f acc = {};
#pragma unroll
  for (int kk = 0; kk < 4; ++kk) {
    v16h a = *(const v16h*)(ht + row * HH + kk * 32 + koff);
    v16h b = *(const v16h*)(W3p + row * HH + kk * 32 + koff);  // col = row
    acc = wmma_f16(a, b, acc);
  }
  return acc;
}

__global__ __launch_bounds__(256) void mlp_step_kernel(
    const float* __restrict__ v, float* __restrict__ s,
    const float* __restrict__ Is,
    const _Float16* __restrict__ W1f, const float* __restrict__ fb1,
    const _Float16* __restrict__ fW2T, const float* __restrict__ fb2,
    const float* __restrict__ fb3,
    const _Float16* __restrict__ W1i, const float* __restrict__ ib1,
    const _Float16* __restrict__ iW2T, const float* __restrict__ ib2,
    const float* __restrict__ ib3,
    const _Float16* __restrict__ W3p,
    _Float16* __restrict__ uT) {
  __shared__ __align__(32) _Float16 hA[8][16 * HH];  // 32 KB activation tiles
  __shared__ float xb[8][16][4];                     // (vn, s0, s1, s2)

  const int tid  = threadIdx.x;
  const int wid  = tid >> 5;
  const int lane = tid & 31;
  const int r0   = blockIdx.x * 128 + wid * 16;
  const int row  = lane & 15;
  const int mb   = (lane >> 4) * 8;
  _Float16* ht   = hA[wid];

  if (lane < 16) {
    int g = r0 + lane;
    xb[wid][lane][0] = (v[g] - VMEAN) / VSTD;
    xb[wid][lane][1] = s[g * 3 + 0];
    xb[wid][lane][2] = s[g * 3 + 1];
    xb[wid][lane][3] = s[g * 3 + 2];
  }

  // ---- forcing MLP + state update ----
  mlp_half(wid, lane, ht, xb, W1f, fb1, fW2T, fb2);
  v8f f3 = layer3_wmma(lane, ht, W3p);
  if (row < SS) {                    // cols 0..2: ds/dt for 8 rows each
    float bias = fb3[row];
#pragma unroll
    for (int i = 0; i < 8; ++i) {
      int r = mb + i;
      float snew = xb[wid][r][1 + row] + DT_C * (f3[i] + bias);
      s[(size_t)(r0 + r) * SS + row] = snew;
      xb[wid][r][1 + row] = snew;
    }
  }

  // ---- ionic MLP + u ----
  mlp_half(wid, lane, ht, xb, W1i, ib1, iW2T, ib2);
  v8f g3 = layer3_wmma(lane, ht, W3p);
  if (row == SS) {                   // col 3: ionic head for 8 rows
    float bias = ib3[0];
#pragma unroll
    for (int i = 0; i < 8; ++i) {
      int g = r0 + mb + i;
      float I_ion = (g3[i] + bias) * VSTD + VMEAN;
      float u = v[g] + DT_C * Is[g] - DT_C * I_ion;
      uT[(size_t)(g & 63) * NN + (g >> 6)] = (_Float16)u;  // [B][N] layout
    }
  }
}

// ---------------------------------------------------------------------------
// Host orchestration (graph-capture safe; everything on `stream`).
// Workspace (~105 MB): invH | massT | P | vbuf | sbuf | fW2T | iW2T |
//                      W1f | W1i | W3p | uT
// ---------------------------------------------------------------------------
extern "C" void kernel_launch(void* const* d_in, const int* in_sizes, int n_in,
                              void* d_out, int out_size, void* d_ws, size_t ws_size,
                              hipStream_t stream) {
  (void)in_sizes; (void)n_in; (void)out_size; (void)ws_size;
  const float* v0   = (const float*)d_in[0];
  const float* s0   = (const float*)d_in[1];
  const float* Isal = (const float*)d_in[2];
  const float* mass = (const float*)d_in[3];
  const float* ipde = (const float*)d_in[4];
  const float* fW1 = (const float*)d_in[5];  const float* fb1 = (const float*)d_in[6];
  const float* fW2 = (const float*)d_in[7];  const float* fb2 = (const float*)d_in[8];
  const float* fW3 = (const float*)d_in[9];  const float* fb3 = (const float*)d_in[10];
  const float* iW1 = (const float*)d_in[11]; const float* ib1 = (const float*)d_in[12];
  const float* iW2 = (const float*)d_in[13]; const float* ib2 = (const float*)d_in[14];
  const float* iW3 = (const float*)d_in[15]; const float* ib3 = (const float*)d_in[16];

  char* w = (char*)d_ws;
  size_t off = 0;
  auto alloc = [&](size_t bytes) -> void* {
    void* p = w + off;
    off += (bytes + 255) & ~(size_t)255;
    return p;
  };
  _Float16* invH  = (_Float16*)alloc((size_t)NN * NN * 2);
  _Float16* massT = (_Float16*)alloc((size_t)NN * NN * 2);
  _Float16* P     = (_Float16*)alloc((size_t)NN * NN * 2);
  float*    vbuf  = (float*)   alloc((size_t)NN * BB * 4);
  float*    sbuf  = (float*)   alloc((size_t)NN * BB * SS * 4);
  _Float16* fW2Th = (_Float16*)alloc((size_t)HH * HH * 2);
  _Float16* iW2Th = (_Float16*)alloc((size_t)HH * HH * 2);
  _Float16* W1fp  = (_Float16*)alloc((size_t)HH * 32 * 2);
  _Float16* W1ip  = (_Float16*)alloc((size_t)HH * 32 * 2);
  _Float16* W3p   = (_Float16*)alloc((size_t)16 * HH * 2);
  _Float16* uT    = (_Float16*)alloc((size_t)NN * BB * 2);

  // One-time conversions / packing
  cvt_f16  <<<(NN * NN) / 256, 256, 0, stream>>>(ipde, invH, NN * NN);
  cvt_f16_T<<<(NN * NN) / 256, 256, 0, stream>>>(mass, massT, NN, NN);
  cvt_f16_T<<<(HH * HH + 255) / 256, 256, 0, stream>>>(fW2, fW2Th, HH, HH);
  cvt_f16_T<<<(HH * HH + 255) / 256, 256, 0, stream>>>(iW2, iW2Th, HH, HH);
  pack_w1<<<(HH * 32) / 128, 128, 0, stream>>>(fW1, W1fp);
  pack_w1<<<(HH * 32) / 128, 128, 0, stream>>>(iW1, W1ip);
  pack_w3<<<(16 * HH) / 128, 128, 0, stream>>>(fW3, iW3, W3p);

  // One-time: P = inv_pde @ mass
  gemm_wmma<4><<<((NN / 16) * (NN / 64)) / 8, 256, 0, stream>>>(
      invH, massT, nullptr, P, NN, NN, NN);

  (void)hipMemcpyAsync(vbuf, v0, (size_t)NN * BB * 4,
                       hipMemcpyDeviceToDevice, stream);
  (void)hipMemcpyAsync(sbuf, s0, (size_t)NN * BB * SS * 4,
                       hipMemcpyDeviceToDevice, stream);

  for (int t = 0; t < TT; ++t) {
    mlp_step_kernel<<<(NN * BB) / 128, 256, 0, stream>>>(
        vbuf, sbuf, Isal + (size_t)t * NN * BB,
        W1fp, fb1, fW2Th, fb2, fb3,
        W1ip, ib1, iW2Th, ib2, ib3, W3p, uT);
    float* vout = (t == TT - 1) ? (float*)d_out : vbuf;
    gemm_step_tdm<<<NN / 128, 256, 0, stream>>>(P, uT, vout);
  }
}
